// minivnGPT_75316546503261
// MI455X (gfx1250) — matrix operand order repaired and checked by reading, not compiled, not run on hardware
//
#include <hip/hip_runtime.h>
#include <stdint.h>
#include <stddef.h>

#define DEV __device__ __forceinline__

typedef __attribute__((ext_vector_type(16))) __bf16    v16bf;
typedef __attribute__((ext_vector_type(8)))  float     v8f;
typedef __attribute__((ext_vector_type(4)))  uint32_t  v4u;
typedef __attribute__((ext_vector_type(8)))  int       v8i;
typedef __attribute__((ext_vector_type(4)))  int       v4i;

union FragBF { uint32_t u[8]; v16bf v; };
union FragF  { float    f[8]; v8f   v; };

constexpr int Bn = 4, Tn = 1024, Dn = 1024, Hn = 16, Ln = 8, DKn = 64, Vn = 32000;
constexpr int BT = Bn * Tn;   // 4096 token rows

#if __has_builtin(__builtin_amdgcn_tensor_load_to_lds) && __has_builtin(__builtin_amdgcn_s_wait_tensorcnt)
#define HAVE_TDM 1
#else
#define HAVE_TDM 0
#endif

DEV uint16_t f2bf(float f) {
  uint32_t u = __builtin_bit_cast(uint32_t, f);
  u += 0x7FFFu + ((u >> 16) & 1u);      // round-to-nearest-even
  return (uint16_t)(u >> 16);
}

DEV v8f vzero8() {
  v8f z;
#pragma unroll
  for (int i = 0; i < 8; i++) z[i] = 0.0f;
  return z;
}

#if HAVE_TDM
// TDM: load a 128-row x 32-col bf16 tile (row stride K elements) into LDS,
// with 1-dword padding after every 16 dwords -> LDS layout [128][17] dwords.
// 6-arg builtin form (clang-23 / therock-10.0): (v4u, v8i, v4i, v4i, v8i, i32 cpol)
DEV void tdm_load_A_tile(const uint16_t* gptr, uint32_t lds_off, int K) {
  uint64_t ga = (uint64_t)(uintptr_t)gptr;
  v4u g0;
  g0[0] = 1u;                                               // count=1, user D#
  g0[1] = lds_off;                                          // lds_addr
  g0[2] = (uint32_t)ga;                                     // global_addr lo
  g0[3] = (uint32_t)((ga >> 32) & 0x01FFFFFFu) | (2u << 30);// ga hi | type=2
  const uint32_t d0 = (uint32_t)K;      // tensor_dim0
  const uint32_t d1 = 1u << 20;         // tensor_dim1 (large, no OOB)
  v8i g1;
  g1[0] = (int)((1u << 16) | (1u << 20) | (3u << 22));      // 2B elems, pad 1dw/16dw
  g1[1] = (int)((d0 & 0xFFFFu) << 16);
  g1[2] = (int)((d0 >> 16) | ((d1 & 0xFFFFu) << 16));
  g1[3] = (int)((d1 >> 16) | (32u << 16));                  // tile_dim0 = 32
  g1[4] = (int)128;                                         // tile_dim1 = 128
  g1[5] = (int)(uint32_t)K;                                 // dim0 stride
  g1[6] = 0;
  g1[7] = 0;
  v4i g2; g2[0] = 1; g2[1] = 1; g2[2] = 0; g2[3] = 0;       // dims 2/3 = 1
  v4i g3; g3[0] = 0; g3[1] = 0; g3[2] = 0; g3[3] = 0;
  v8i g4;
#pragma unroll
  for (int i = 0; i < 8; i++) g4[i] = 0;
  __builtin_amdgcn_tensor_load_to_lds(g0, g1, g2, g3, g4, 0);
}
#endif

// ---------------------------------------------------------------------------
// Elementwise fp32 -> bf16 weight conversion
// ---------------------------------------------------------------------------
__global__ __launch_bounds__(256) void cvt_kernel(const float* __restrict__ s,
                                                  uint16_t* __restrict__ d, size_t n) {
  size_t i = (size_t)blockIdx.x * 256 + threadIdx.x;
  size_t stride = (size_t)gridDim.x * 256;
  for (; i < n; i += stride) d[i] = f2bf(s[i]);
}

// Wq/Wk/Wv: (L,H,D,DK) -> bf16 (L, D, H*DK)  so QKV are plain DxD GEMMs
__global__ __launch_bounds__(256) void cvt_wqkv_kernel(const float* __restrict__ s,
                                                       uint16_t* __restrict__ d) {
  const size_t total = (size_t)Ln * Dn * Dn;
  size_t i = (size_t)blockIdx.x * 256 + threadIdx.x;
  size_t stride = (size_t)gridDim.x * 256;
  for (; i < total; i += stride) {
    int col  = (int)(i & (Dn - 1));        // h*DK + kk
    size_t r = i >> 10;
    int drow = (int)(r & (Dn - 1));
    int l    = (int)(r >> 10);
    int hh   = col >> 6;                   // / DK
    int kk   = col & 63;
    d[i] = f2bf(s[(((size_t)(l * Hn + hh) * Dn + drow) << 6) + kk]);
  }
}

// ---------------------------------------------------------------------------
// Embedding gather
// ---------------------------------------------------------------------------
__global__ __launch_bounds__(256) void embed_kernel(const int* __restrict__ tok,
                                                    const float* __restrict__ emb,
                                                    float* __restrict__ x) {
  int row = blockIdx.x;
  int t = tok[row];
  const float4* s = (const float4*)(emb + (size_t)t * Dn);
  float4* d = (float4*)(x + (size_t)row * Dn);
  for (int i = threadIdx.x; i < Dn / 4; i += 256) d[i] = s[i];
}

// ---------------------------------------------------------------------------
// LayerNorm: fp32 in -> bf16 out (one block per row)
// ---------------------------------------------------------------------------
__global__ __launch_bounds__(256) void layernorm_kernel(const float* __restrict__ x,
                                                        const float* __restrict__ g,
                                                        const float* __restrict__ bb,
                                                        uint16_t* __restrict__ out) {
  const int row = blockIdx.x;
  const float* xr = x + (size_t)row * Dn;
  float s = 0.f, s2 = 0.f;
  for (int i = threadIdx.x; i < Dn; i += 256) { float v = xr[i]; s += v; s2 += v * v; }
#pragma unroll
  for (int off = 16; off > 0; off >>= 1) { s += __shfl_xor(s, off); s2 += __shfl_xor(s2, off); }
  __shared__ float rs[8], rs2[8];
  __shared__ float fm, fr;
  const int w = threadIdx.x >> 5, ln = threadIdx.x & 31;
  if (ln == 0) { rs[w] = s; rs2[w] = s2; }
  __syncthreads();
  if (threadIdx.x == 0) {
    float S = 0.f, S2 = 0.f;
#pragma unroll
    for (int i = 0; i < 8; i++) { S += rs[i]; S2 += rs2[i]; }
    float mean = S / Dn;
    float var  = S2 / Dn - mean * mean;
    fm = mean; fr = rsqrtf(var + 1e-5f);
  }
  __syncthreads();
  const float mean = fm, rstd = fr;
  uint16_t* o = out + (size_t)row * Dn;
  for (int i = threadIdx.x; i < Dn; i += 256)
    o[i] = f2bf((xr[i] - mean) * rstd * g[i] + bb[i]);
}

// ---------------------------------------------------------------------------
// RoPE on q,k (fp32 in, bf16 out). Layout (B,T,H,DK); pair index = flat/2.
// ---------------------------------------------------------------------------
__global__ __launch_bounds__(256) void rope_kernel(const float* __restrict__ q,
                                                   const float* __restrict__ k,
                                                   uint16_t* __restrict__ qo,
                                                   uint16_t* __restrict__ ko) {
  size_t idx = (size_t)blockIdx.x * 256 + threadIdx.x;      // pair index
  int i      = (int)(idx & (DKn / 2 - 1));                  // 0..31
  size_t rest = idx >> 5;                                   // bt*H + h
  size_t bt   = rest >> 4;
  int t = (int)(bt & (Tn - 1));
  float inv = expf(-((float)(2 * i) / (float)DKn) * 9.210340371976184f); // ln(1e4)
  float ang = (float)t * inv;
  float c = cosf(ang), s = sinf(ang);
  float a0 = q[2 * idx], b0 = q[2 * idx + 1];
  qo[2 * idx]     = f2bf(a0 * c - b0 * s);
  qo[2 * idx + 1] = f2bf(a0 * s + b0 * c);
  float a1 = k[2 * idx], b1 = k[2 * idx + 1];
  ko[2 * idx]     = f2bf(a1 * c - b1 * s);
  ko[2 * idx + 1] = f2bf(a1 * s + b1 * c);
}

// ---------------------------------------------------------------------------
// WMMA GEMM: out[M,N] = A(bf16,MxK) @ B(bf16,KxN) (+resid)(+bias)
// 128x128 block tile, BK=32, 4 waves, wave tile 64x64 (4x4 = 16 WMMA accs).
// Double-buffered LDS; A tile via TDM (hardware pad -> [128][17] dwords),
// B tile via pair-interleaved manual stage. Branch-free templated epilogue.
// Uniform branches forced scalar via readfirstlane (EXEC stays all-ones).
// ---------------------------------------------------------------------------
template <bool WF, bool WB, bool RES, bool BIAS>
__global__ __launch_bounds__(128) void gemm_bf16_t(const uint16_t* __restrict__ A,
                                                   const uint16_t* __restrict__ Bw,
                                                   float* __restrict__ outF,
                                                   uint16_t* __restrict__ outB,
                                                   const float* __restrict__ resid,
                                                   const float* __restrict__ bias,
                                                   int N, int K) {
  __shared__ uint32_t As[2][128][17];
  __shared__ uint32_t Bs[2][16][128];
  const int tid  = threadIdx.x;
  const int lane = tid & 31, wave = tid >> 5;
  const int hsel = lane >> 4, l16 = lane & 15;
  const int m0b = blockIdx.y * 128;
  const int n0b = blockIdx.x * 128;
  const int wm0 = (wave >> 1) * 64;
  const int wn0 = (wave & 1) * 64;
  const int waveS = __builtin_amdgcn_readfirstlane(wave);   // scalar wave id

  // per-thread B stage coords: 2 k-pair rows x 8 n, two sub-chunks (c=0,1)
  const int bkp0 = tid >> 4;           // k-pair 0..7 (+8 for c=1)
  const int bnc  = (tid & 15) * 8;     // n offset

  v8f acc[4][4];
#pragma unroll
  for (int i = 0; i < 4; i++)
#pragma unroll
    for (int j = 0; j < 4; j++) acc[i][j] = vzero8();

  const int nT = K >> 5;

  // ---- prologue: stage tile 0 into buffer 0
#if HAVE_TDM
  if (waveS == 0)
    tdm_load_A_tile(A + (size_t)m0b * K, (uint32_t)(size_t)&As[0][0][0], K);
#else
#pragma unroll
  for (int c = 0; c < 4; c++) {
    int idx = tid + c * 128;
    int row = idx >> 2, kc = (idx & 3) * 8;
    uint4 u = *(const uint4*)(A + (size_t)(m0b + row) * K + kc);
    uint32_t* dst = &As[0][row][kc >> 1];
    dst[0] = u.x; dst[1] = u.y; dst[2] = u.z; dst[3] = u.w;
  }
#endif
#pragma unroll
  for (int c = 0; c < 2; c++) {
    int kp = bkp0 + c * 8;
    const uint32_t* p0 = (const uint32_t*)(Bw + (size_t)(kp * 2) * N + n0b + bnc);
    const uint32_t* p1 = (const uint32_t*)(Bw + (size_t)(kp * 2 + 1) * N + n0b + bnc);
#pragma unroll
    for (int i = 0; i < 4; i++) {
      uint32_t w0 = p0[i], w1 = p1[i];
      Bs[0][kp][bnc + 2 * i]     = (w0 & 0xFFFFu) | (w1 << 16);
      Bs[0][kp][bnc + 2 * i + 1] = (w0 >> 16)     | (w1 & 0xFFFF0000u);
    }
  }
#if HAVE_TDM
  if (waveS == 0) __builtin_amdgcn_s_wait_tensorcnt(0);
#endif
  __syncthreads();

  for (int t = 0; t < nT; t++) {
    const int cb = t & 1, nb = (t + 1) & 1;
    const bool pf = (t + 1) < nT;
    const int ktn = (t + 1) << 5;

    // ---- issue next tile's fetches (overlaps with WMMA below)
    uint32_t bw0[2][4], bw1[2][4];
#if !HAVE_TDM
    uint4 aw[4];
#endif
    if (pf) {
#if HAVE_TDM
      if (waveS == 0)
        tdm_load_A_tile(A + (size_t)m0b * K + ktn, (uint32_t)(size_t)&As[nb][0][0], K);
#else
#pragma unroll
      for (int c = 0; c < 4; c++) {
        int idx = tid + c * 128;
        aw[c] = *(const uint4*)(A + (size_t)(m0b + (idx >> 2)) * K + ktn + (idx & 3) * 8);
      }
#endif
#pragma unroll
      for (int c = 0; c < 2; c++) {
        int kp = bkp0 + c * 8;
        const uint32_t* p0 = (const uint32_t*)(Bw + (size_t)(ktn + kp * 2) * N + n0b + bnc);
        const uint32_t* p1 = (const uint32_t*)(Bw + (size_t)(ktn + kp * 2 + 1) * N + n0b + bnc);
#pragma unroll
        for (int i = 0; i < 4; i++) { bw0[c][i] = p0[i]; bw1[c][i] = p1[i]; }
      }
    }

    // ---- compute current tile: 4 A-frags x 4 B-frags -> 16 WMMAs
    FragBF afr[4], bfr[4];
#pragma unroll
    for (int i = 0; i < 4; i++) {
      int m = wm0 + i * 16 + l16;
#pragma unroll
      for (int v = 0; v < 8; v++)
        afr[i].u[v] = As[cb][m][(v >> 2) * 8 + hsel * 4 + (v & 3)];
    }
#pragma unroll
    for (int j = 0; j < 4; j++) {
      int n = wn0 + j * 16 + l16;
#pragma unroll
      for (int v = 0; v < 8; v++)
        bfr[j].u[v] = Bs[cb][(v >> 2) * 8 + hsel * 4 + (v & 3)][n];
    }
#pragma unroll
    for (int i = 0; i < 4; i++)
#pragma unroll
      for (int j = 0; j < 4; j++)
        acc[i][j] = __builtin_amdgcn_wmma_f32_16x16x32_bf16(
            false, afr[i].v, false, bfr[j].v, (short)0, acc[i][j], false, false);

    // ---- publish next tile
    if (pf) {
#pragma unroll
      for (int c = 0; c < 2; c++) {
        int kp = bkp0 + c * 8;
#pragma unroll
        for (int i = 0; i < 4; i++) {
          Bs[nb][kp][bnc + 2 * i]     = (bw0[c][i] & 0xFFFFu) | (bw1[c][i] << 16);
          Bs[nb][kp][bnc + 2 * i + 1] = (bw0[c][i] >> 16)     | (bw1[c][i] & 0xFFFF0000u);
        }
      }
#if !HAVE_TDM
#pragma unroll
      for (int c = 0; c < 4; c++) {
        int idx = tid + c * 128;
        uint32_t* d = &As[nb][idx >> 2][((idx & 3) * 8) >> 1];
        d[0] = aw[c].x; d[1] = aw[c].y; d[2] = aw[c].z; d[3] = aw[c].w;
      }
#endif
#if HAVE_TDM
      if (waveS == 0) __builtin_amdgcn_s_wait_tensorcnt(0);
#endif
    }
    __syncthreads();
  }

  // ---- branch-free epilogue
#pragma unroll
  for (int i = 0; i < 4; i++)
#pragma unroll
    for (int j = 0; j < 4; j++) {
      FragF cf; cf.v = acc[i][j];
      int n = n0b + wn0 + j * 16 + l16;
      float bv = 0.0f;
      if (BIAS) bv = bias[n];
#pragma unroll
      for (int e = 0; e < 8; e++) {
        int m = m0b + wm0 + i * 16 + hsel * 8 + e;
        size_t idx = (size_t)m * N + n;
        float s = cf.f[e] + bv;
        if (RES) s += resid[idx];
        if (WF) outF[idx] = s;
        if (WB) outB[idx] = f2bf(s);
      }
    }
}

// ---------------------------------------------------------------------------
// Flash attention: grid (T/64, H, B), 128 threads (4 waves x 16 q rows).
// q/k/v bf16 in (B,T,H,DK) layout; output bf16 same layout. Causal.
// ---------------------------------------------------------------------------
__global__ __launch_bounds__(128) void attention_kernel(const uint16_t* __restrict__ qb,
                                                        const uint16_t* __restrict__ kb,
                                                        const uint16_t* __restrict__ vb,
                                                        uint16_t* __restrict__ ob) {
  __shared__ uint32_t Ks[32][33];      // [key][dk/2] pair dwords
  __shared__ uint32_t Vs[16][64];      // [s/2][dk] pair-interleaved
  __shared__ uint32_t Ps[4][16][17];   // per-wave P staging [m][s/2]

  const int tid = threadIdx.x, lane = tid & 31, wave = tid >> 5;
  const int hsel = lane >> 4, l16 = lane & 15;
  const int qt0b = blockIdx.x * 64;
  const int h = blockIdx.y, b = blockIdx.z;
  const int m0 = qt0b + wave * 16;
  const int m0S = __builtin_amdgcn_readfirstlane(m0);   // scalar copy for branches
  const size_t base = ((size_t)b * Tn) * Dn + (size_t)h * DKn;

  // Q fragments (two K=32 windows over DK=64), A-layout
  FragBF qf[2];
  {
    const uint32_t* qrow = (const uint32_t*)(qb + base + (size_t)(m0 + l16) * Dn);
#pragma unroll
    for (int f = 0; f < 2; f++)
#pragma unroll
      for (int v = 0; v < 8; v++)
        qf[f].u[v] = qrow[f * 16 + (v >> 2) * 8 + hsel * 4 + (v & 3)];
  }

  FragF o[4];
#pragma unroll
  for (int a = 0; a < 4; a++) o[a].v = vzero8();
  float rm[8], rl[8];
#pragma unroll
  for (int e = 0; e < 8; e++) { rm[e] = -1e30f; rl[e] = 0.0f; }

  const int nS = blockIdx.x * 2 + 2;   // key blocks of 32 up to block's last row
  for (int sb = 0; sb < nS; sb++) {
    const int s0 = sb * 32;
    __syncthreads();
    // K tile: 32 keys x 64 bf16
    {
      int row = tid >> 2, ch = (tid & 3) * 8;  // 8 dwords per thread
      const uint4* src = (const uint4*)(kb + base + (size_t)(s0 + row) * Dn + ch * 2);
      uint4 u0 = src[0], u1 = src[1];
      uint32_t* dst = &Ks[row][ch];
      dst[0] = u0.x; dst[1] = u0.y; dst[2] = u0.z; dst[3] = u0.w;
      dst[4] = u1.x; dst[5] = u1.y; dst[6] = u1.z; dst[7] = u1.w;
      if (sb + 1 < nS)   // prefetch next key tile (global_prefetch_b8)
        __builtin_prefetch(kb + base + (size_t)(s0 + 32 + row) * Dn + ch * 2, 0, 1);
    }
    // V tile: 32 keys x 64, pair-interleaved along key
    {
      int sp = tid >> 3, dk0 = (tid & 7) * 8;
      const uint32_t* p0 = (const uint32_t*)(vb + base + (size_t)(s0 + 2 * sp) * Dn + dk0);
      const uint32_t* p1 = (const uint32_t*)(vb + base + (size_t)(s0 + 2 * sp + 1) * Dn + dk0);
#pragma unroll
      for (int i = 0; i < 4; i++) {
        uint32_t w0 = p0[i], w1 = p1[i];
        Vs[sp][dk0 + 2 * i]     = (w0 & 0xFFFFu) | (w1 << 16);
        Vs[sp][dk0 + 2 * i + 1] = (w0 >> 16)     | (w1 & 0xFFFF0000u);
      }
      if (sb + 1 < nS)
        __builtin_prefetch(vb + base + (size_t)(s0 + 32 + 2 * sp) * Dn + dk0, 0, 1);
    }
    __syncthreads();

    if (s0 <= m0S) {   // scalar causal skip: EXEC untouched around WMMA
      FragF sc[2];
#pragma unroll
      for (int nt = 0; nt < 2; nt++) {
        v8f a = vzero8();
#pragma unroll
        for (int f = 0; f < 2; f++) {
          FragBF kf;   // B-frag: K^T, lane = key column
#pragma unroll
          for (int v = 0; v < 8; v++)
            kf.u[v] = Ks[nt * 16 + l16][f * 16 + (v >> 2) * 8 + hsel * 4 + (v & 3)];
          a = __builtin_amdgcn_wmma_f32_16x16x32_bf16(
              false, qf[f].v, false, kf.v, (short)0, a, false, false);
        }
        sc[nt].v = a;
      }
      // online softmax over this 32-key block
      uint16_t* Pw = (uint16_t*)&Ps[wave][0][0];
#pragma unroll
      for (int e = 0; e < 8; e++) {
        int mrow = e + hsel * 8;
        int trow = m0 + mrow;
        float v0 = sc[0].f[e] * 0.125f;   // DK^-0.5
        float v1 = sc[1].f[e] * 0.125f;
        if (s0 + l16 > trow)      v0 = -1e30f;
        if (s0 + 16 + l16 > trow) v1 = -1e30f;
        float mx = fmaxf(v0, v1);
        mx = fmaxf(mx, __shfl_xor(mx, 1));
        mx = fmaxf(mx, __shfl_xor(mx, 2));
        mx = fmaxf(mx, __shfl_xor(mx, 4));
        mx = fmaxf(mx, __shfl_xor(mx, 8));
        float nm = fmaxf(rm[e], mx);
        float alpha = __expf(rm[e] - nm);
        float p0 = __expf(v0 - nm), p1 = __expf(v1 - nm);
        float ps = p0 + p1;
        ps += __shfl_xor(ps, 1); ps += __shfl_xor(ps, 2);
        ps += __shfl_xor(ps, 4); ps += __shfl_xor(ps, 8);
        rl[e] = rl[e] * alpha + ps;
        rm[e] = nm;
#pragma unroll
        for (int a2 = 0; a2 < 4; a2++) o[a2].f[e] *= alpha;
        Pw[mrow * 34 + l16]      = f2bf(p0);
        Pw[mrow * 34 + 16 + l16] = f2bf(p1);
      }
      // P(16x32) @ V(32x64)
      FragBF pf;
#pragma unroll
      for (int v = 0; v < 8; v++)
        pf.u[v] = Ps[wave][l16][(v >> 2) * 8 + hsel * 4 + (v & 3)];
#pragma unroll
      for (int a2 = 0; a2 < 4; a2++) {
        FragBF vf;
#pragma unroll
        for (int v = 0; v < 8; v++)
          vf.u[v] = Vs[(v >> 2) * 8 + hsel * 4 + (v & 3)][a2 * 16 + l16];
        o[a2].v = __builtin_amdgcn_wmma_f32_16x16x32_bf16(
            false, pf.v, false, vf.v, (short)0, o[a2].v, false, false);
      }
    }
  }

#pragma unroll
  for (int e = 0; e < 8; e++) rl[e] = 1.0f / rl[e];
#pragma unroll
  for (int a2 = 0; a2 < 4; a2++)
#pragma unroll
    for (int e = 0; e < 8; e++) {
      int trow = m0 + e + hsel * 8;
      ob[base + (size_t)trow * Dn + a2 * 16 + l16] = f2bf(o[a2].f[e] * rl[e]);
    }
}

// ---------------------------------------------------------------------------
// Host orchestration
// ---------------------------------------------------------------------------
extern "C" void kernel_launch(void* const* d_in, const int* in_sizes, int n_in,
                              void* d_out, int out_size, void* d_ws, size_t ws_size,
                              hipStream_t stream) {
  (void)in_sizes; (void)n_in; (void)out_size; (void)ws_size;
  const int*   tokens = (const int*)  d_in[0];
  const float* emb    = (const float*)d_in[1];
  const float* Wq     = (const float*)d_in[2];
  const float* Wk     = (const float*)d_in[3];
  const float* Wv     = (const float*)d_in[4];
  const float* Wproj  = (const float*)d_in[5];
  const float* bproj  = (const float*)d_in[6];
  const float* Wffn   = (const float*)d_in[7];
  const float* bffn   = (const float*)d_in[8];
  const float* ln1g   = (const float*)d_in[9];
  const float* ln1b   = (const float*)d_in[10];
  const float* ln2g   = (const float*)d_in[11];
  const float* ln2b   = (const float*)d_in[12];
  const float* lnfg   = (const float*)d_in[13];
  const float* lnfb   = (const float*)d_in[14];
  const float* Wout   = (const float*)d_in[15];
  const float* bout   = (const float*)d_in[16];
  float* logits = (float*)d_out;

  char* ws = (char*)d_ws;
  size_t off = 0;
  auto carve = [&](size_t bytes) -> char* {
    char* p = ws + off;
    off += (bytes + 255) & ~(size_t)255;
    return p;
  };
  float*    x    = (float*)   carve((size_t)BT * Dn * 4);
  uint16_t* hbf  = (uint16_t*)carve((size_t)BT * Dn * 2);
  float*    qf32 = (float*)   carve((size_t)BT * Dn * 4);
  float*    kf32 = (float*)   carve((size_t)BT * Dn * 4);
  uint16_t* qbf  = (uint16_t*)carve((size_t)BT * Dn * 2);
  uint16_t* kbf  = (uint16_t*)carve((size_t)BT * Dn * 2);
  uint16_t* vbf  = (uint16_t*)carve((size_t)BT * Dn * 2);
  uint16_t* abf  = (uint16_t*)carve((size_t)BT * Dn * 2);
  uint16_t* wqb  = (uint16_t*)carve((size_t)Ln * Dn * Dn * 2);
  uint16_t* wkb  = (uint16_t*)carve((size_t)Ln * Dn * Dn * 2);
  uint16_t* wvb  = (uint16_t*)carve((size_t)Ln * Dn * Dn * 2);
  uint16_t* wpb  = (uint16_t*)carve((size_t)Ln * Dn * Dn * 2);
  uint16_t* wfb  = (uint16_t*)carve((size_t)Ln * Dn * Dn * 2);
  uint16_t* wob  = (uint16_t*)carve((size_t)Dn * Vn * 2);

  cvt_wqkv_kernel<<<4096, 256, 0, stream>>>(Wq, wqb);
  cvt_wqkv_kernel<<<4096, 256, 0, stream>>>(Wk, wkb);
  cvt_wqkv_kernel<<<4096, 256, 0, stream>>>(Wv, wvb);
  cvt_kernel<<<4096, 256, 0, stream>>>(Wproj, wpb, (size_t)Ln * Dn * Dn);
  cvt_kernel<<<4096, 256, 0, stream>>>(Wffn,  wfb, (size_t)Ln * Dn * Dn);
  cvt_kernel<<<8192, 256, 0, stream>>>(Wout,  wob, (size_t)Dn * Vn);
  embed_kernel<<<BT, 256, 0, stream>>>(tokens, emb, x);

  dim3 gD(Dn / 128, BT / 128);   // 8 x 32
  for (int l = 0; l < Ln; l++) {
    size_t wo = (size_t)l * Dn * Dn;
    layernorm_kernel<<<BT, 256, 0, stream>>>(x, ln1g + l * Dn, ln1b + l * Dn, hbf);
    gemm_bf16_t<true, false, false, false><<<gD, 128, 0, stream>>>(
        hbf, wqb + wo, qf32, nullptr, nullptr, nullptr, Dn, Dn);
    gemm_bf16_t<true, false, false, false><<<gD, 128, 0, stream>>>(
        hbf, wkb + wo, kf32, nullptr, nullptr, nullptr, Dn, Dn);
    gemm_bf16_t<false, true, false, false><<<gD, 128, 0, stream>>>(
        hbf, wvb + wo, nullptr, vbf, nullptr, nullptr, Dn, Dn);
    rope_kernel<<<(BT * Dn / 2) / 256, 256, 0, stream>>>(qf32, kf32, qbf, kbf);
    attention_kernel<<<dim3(Tn / 64, Hn, Bn), 128, 0, stream>>>(qbf, kbf, vbf, abf);
    gemm_bf16_t<true, false, true, true><<<gD, 128, 0, stream>>>(
        abf, wpb + wo, x, nullptr, x, bproj + l * Dn, Dn, Dn);
    layernorm_kernel<<<BT, 256, 0, stream>>>(x, ln2g + l * Dn, ln2b + l * Dn, hbf);
    gemm_bf16_t<true, false, true, true><<<gD, 128, 0, stream>>>(
        hbf, wfb + wo, x, nullptr, x, bffn + l * Dn, Dn, Dn);
  }
  layernorm_kernel<<<BT, 256, 0, stream>>>(x, lnfg, lnfb, hbf);
  gemm_bf16_t<true, false, false, true><<<dim3(Vn / 128, BT / 128), 128, 0, stream>>>(
      hbf, wob, logits, nullptr, nullptr, bout, Vn, Dn);
}
